// LengthRegulator_23596550324482
// MI455X (gfx1250) — matrix operand order repaired and checked by reading, not compiled, not run on hardware
//
#include <hip/hip_runtime.h>

typedef __attribute__((ext_vector_type(16))) _Float16 v16h;
typedef __attribute__((ext_vector_type(8)))  _Float16 v8h;
typedef __attribute__((ext_vector_type(8)))  float    v8f;

#define EMB     384
#define GROUPS  4
#define GC      96          // channels per group
#define BATCH   32
#define T_IN    1024
#define TP      1025        // T+1 rows after silence concat
#define KDIM    288         // 3 taps * 96 in-channels per group
#define KSTEPS  9           // 288 / 32
#define NTILES  6           // 96 / 16 output-channel tiles per group
#define MTILES  65          // ceil(1025/16)

// workspace layout
#define BPACK_HALFS   (GROUPS * NTILES * KSTEPS * 32 * 16)   // 110592 halfs
#define CUM_OFF_BYTES (BPACK_HALFS * 2)                      // 221184 bytes

// ---------------------------------------------------------------------------
// Repack conv1_w [384][96][3] f32 -> f16 B-fragments in exact WMMA lane layout.
// Fragment element j of lane L for k-step c maps to K = c*32 + (j/8)*16 + (L/16)*8 + (j%8),
// output channel o = g*96 + n*16 + (L%16).  K decomposes as K = ktap*96 + i.
// ---------------------------------------------------------------------------
__global__ void pack_weights_kernel(const float* __restrict__ w1,
                                    _Float16* __restrict__ bpack) {
    int f = blockIdx.x * 256 + threadIdx.x;
    if (f >= BPACK_HALFS) return;
    int j    = f & 15;
    int lane = (f >> 4) & 31;
    int c    = (f >> 9) % KSTEPS;
    int n    = (f / (KSTEPS * 512)) % NTILES;
    int g    = f / (NTILES * KSTEPS * 512);
    int K    = c * 32 + (j >> 3) * 16 + (lane >> 4) * 8 + (j & 7);
    int ktap = K / GC;
    int i    = K - ktap * GC;
    int o    = g * GC + n * 16 + (lane & 15);
    bpack[f] = (_Float16)w1[(o * GC + i) * 3 + ktap];
}

// ---------------------------------------------------------------------------
// Per-batch inclusive cumsum of durations [32][1025] -> cum (int32) in ws.
// One wave per batch; shfl-based scan, 33 chunks of 32.
// ---------------------------------------------------------------------------
__global__ void cumsum_kernel(const int* __restrict__ dur, int* __restrict__ cum) {
    int b = blockIdx.x;
    int lane = threadIdx.x;
    int running = 0;
    for (int base = 0; base < TP; base += 32) {
        int idx = base + lane;
        int v = (idx < TP) ? dur[b * TP + idx] : 0;
        #pragma unroll
        for (int d = 1; d < 32; d <<= 1) {
            int nb = __shfl_up(v, d, 32);
            if (lane >= d) v += nb;
        }
        v += running;
        if (idx < TP) cum[b * TP + idx] = v;
        running = __shfl(v, 31, 32);
    }
}

// ---------------------------------------------------------------------------
// ph row loader with silence concat + conv zero padding.
// ---------------------------------------------------------------------------
__device__ __forceinline__ float load_ph(const float* __restrict__ ph,
                                         const float* __restrict__ sil,
                                         int b, int t, int ch) {
    if (t < 0 || t >= TP) return 0.0f;
    if (t == T_IN) return sil[ch];
    return ph[((size_t)b * T_IN + t) * EMB + ch];
}

// ---------------------------------------------------------------------------
// Duration predictor: grouped conv1 (as 16x16x32 f16 WMMA GEMM, f32 acc)
// -> ReLU(+bias) -> 1x1 conv reduction -> dpred[b][t].
// Block = 768 threads = 24 waves = 4 groups x 6 N-tiles; one (b, 16-row M-tile)
// per block. A panel (4 groups x 16 rows x 288 K, f16) staged in LDS.
// ---------------------------------------------------------------------------
__global__ void __launch_bounds__(768)
duration_predictor_kernel(const float* __restrict__ ph,
                          const float* __restrict__ sil,
                          const _Float16* __restrict__ bpack,
                          const float* __restrict__ b1,
                          const float* __restrict__ w2,
                          const float* __restrict__ b2,
                          float* __restrict__ dpred) {
    __shared__ _Float16 Alds[GROUPS * 16 * KDIM];   // 36864 B
    __shared__ float red[16];

    int tid   = threadIdx.x;
    int mtile = blockIdx.x;
    int b     = blockIdx.y;
    int m0    = mtile * 16;

    if (tid < 16) red[tid] = 0.0f;

    // Stage A: 4 groups * 16 rows * 288 K values, f32 -> f16.
    // Consecutive tids read consecutive channels (coalesced).
    #pragma unroll 4
    for (int s = 0; s < 24; ++s) {
        int e    = tid + s * 768;        // 0 .. 18431
        int K    = e % KDIM;
        int rowg = e / KDIM;             // g*16 + row
        int sg   = rowg >> 4;
        int row  = rowg & 15;
        int ktap = K / GC;
        int i    = K - ktap * GC;
        int t    = m0 + row + ktap - 1;
        float v  = load_ph(ph, sil, b, t, sg * GC + i);
        Alds[(sg * 16 + row) * KDIM + K] = (_Float16)v;
    }
    __syncthreads();

    int wave = tid >> 5;        // 0..23
    int lane = tid & 31;
    int g    = wave / NTILES;
    int n    = wave % NTILES;
    int col  = lane & 15;       // N index (and A-fragment M row index)
    int half = lane >> 4;

    // A fragment (ISA layout): lane holds row M = lane%16;
    // VGPR v, sub s -> K = (v&3)*2+s + (v>>2)*16 + half*8  => two contiguous
    // 8-half chunks at K = kk + half*8 and K = kk + half*8 + 16.
    const _Float16* abase = &Alds[(g * 16 + col) * KDIM + half * 8];
    const _Float16* bbase = bpack + (((size_t)(g * NTILES + n) * KSTEPS) * 32 + lane) * 16;

    v8f acc = {};
    #pragma unroll
    for (int c = 0; c < KSTEPS; ++c) {
        const _Float16* ap = abase + c * 32;
        v8h a0 = *(const v8h*)(ap);         // ds_load_b128
        v8h a1 = *(const v8h*)(ap + 16);    // ds_load_b128
        v16h a = __builtin_shufflevector(a0, a1,
                 0, 1, 2, 3, 4, 5, 6, 7, 8, 9, 10, 11, 12, 13, 14, 15);
        v16h bf = *(const v16h*)(bbase + c * 512);  // contiguous 32B per lane
        acc = __builtin_amdgcn_wmma_f32_16x16x32_f16(
                  false, a, false, bf, (short)0, acc, false, false);
    }

    // Epilogue: h = relu(acc + bias); partial dpred = sum_o h * w2[o].
    // C layout: VGPR r holds M = r + 8*half, column N = col.
    int   o    = g * GC + n * 16 + col;
    float bias = b1[o];
    float wv   = w2[o];
    #pragma unroll
    for (int r = 0; r < 8; ++r) {
        float h = acc[r] + bias;
        h = h > 0.0f ? h : 0.0f;
        float sum = h * wv;
        #pragma unroll
        for (int m = 1; m < 16; m <<= 1)
            sum += __shfl_xor(sum, m, 32);   // reduce across 16 columns
        if (col == 0) atomicAdd(&red[r + half * 8], sum);
    }
    __syncthreads();

    if (tid < 16) {
        int t = m0 + tid;
        if (t < TP) dpred[b * TP + t] = red[tid] + b2[0];
    }
}

// ---------------------------------------------------------------------------
// Length regulation: per output frame, binary-search cumsum, copy 384 floats.
// Block = 256 threads = 8 waves = 8 frames; cumsum panel cached in LDS.
// ---------------------------------------------------------------------------
__global__ void __launch_bounds__(256)
regulate_kernel(const float* __restrict__ ph,
                const float* __restrict__ sil,
                const int* __restrict__ cum,
                float* __restrict__ out, int t_out) {
    __shared__ int cum_s[TP];
    int b   = blockIdx.y;
    int tid = threadIdx.x;
    for (int i = tid; i < TP; i += 256) cum_s[i] = cum[b * TP + i];
    __syncthreads();

    int frame = blockIdx.x * 8 + (tid >> 5);
    int lane  = tid & 31;
    if (frame >= t_out) return;

    int total = cum_s[TP - 1];
    bool valid = frame < total;
    // searchsorted(cum, frame, side='right'): first j with cum[j] > frame
    int lo = 0, hi = TP;
    while (lo < hi) {
        int mid = (lo + hi) >> 1;
        if (cum_s[mid] <= frame) lo = mid + 1; else hi = mid;
    }
    int idx = lo > T_IN ? T_IN : lo;   // clip to TP-1

    const float* srow = (idx < T_IN) ? (ph + ((size_t)b * T_IN + idx) * EMB) : sil;
    float4* dst = (float4*)(out + ((size_t)b * t_out + frame) * EMB);
    float4 z = make_float4(0.f, 0.f, 0.f, 0.f);
    #pragma unroll
    for (int k = 0; k < 3; ++k) {
        int e = lane + k * 32;          // 96 float4 per row
        float4 v = valid ? ((const float4*)srow)[e] : z;
        dst[e] = v;
    }
}

// ---------------------------------------------------------------------------
extern "C" void kernel_launch(void* const* d_in, const int* in_sizes, int n_in,
                              void* d_out, int out_size, void* d_ws, size_t ws_size,
                              hipStream_t stream) {
    const float* phoneme  = (const float*)d_in[0];
    const float* silence  = (const float*)d_in[1];
    const float* conv1_w  = (const float*)d_in[2];
    const float* conv1_b  = (const float*)d_in[3];
    const float* conv2_w  = (const float*)d_in[4];
    const float* conv2_b  = (const float*)d_in[5];
    const int*   durations = (const int*)d_in[6];
    (void)in_sizes; (void)n_in; (void)ws_size;

    // out = [expanded: B*t_out*EMB | dpred: B*TP]; recover t_out on host.
    int t_out = (out_size - BATCH * TP) / (BATCH * EMB);
    float* out_exp = (float*)d_out;
    float* dpred   = (float*)d_out + (size_t)BATCH * t_out * EMB;

    _Float16* bpack = (_Float16*)d_ws;
    int*      cum   = (int*)((char*)d_ws + CUM_OFF_BYTES);

    pack_weights_kernel<<<(BPACK_HALFS + 255) / 256, 256, 0, stream>>>(conv1_w, bpack);
    cumsum_kernel<<<BATCH, 32, 0, stream>>>(durations, cum);

    dim3 gg(MTILES, BATCH);
    duration_predictor_kernel<<<gg, 768, 0, stream>>>(
        phoneme, silence, bpack, conv1_b, conv2_w, conv2_b, dpred);

    dim3 gr((t_out + 7) / 8, BATCH);
    regulate_kernel<<<gr, 256, 0, stream>>>(phoneme, silence, cum, out_exp, t_out);
}
